// LSTM_43379169689827
// MI455X (gfx1250) — matrix operand order repaired and checked by reading, not compiled, not run on hardware
//
#include <hip/hip_runtime.h>

#define LAYERS 2
#define SEQ    256
#define BATCH  128
#define INDIM  256
#define HDIM   256
#define G4     (4 * HDIM)   // 1024
#define EPSV   1e-5f

typedef __attribute__((ext_vector_type(16))) __bf16 v16bf;
typedef __attribute__((ext_vector_type(8)))  float  v8f;

union FragU { unsigned int u[8]; v16bf v; };

// A-matrix 16x32 bf16 fragment, per CDNA5 ISA 7.12.2 16-bit A layout:
// lane L: m = L%16, khalf = L/16; VGPR i: grp=i/4, j=i%4, k0 = grp*16 + khalf*8 + 2j
__device__ __forceinline__ v16bf load_frag_a(const __bf16* A, int lda, int lane) {
  FragU f;
  const int m = lane & 15, kh = lane >> 4;
#pragma unroll
  for (int i = 0; i < 8; ++i) {
    const int grp = i >> 2, j = i & 3;
    const int k0 = grp * 16 + kh * 8 + 2 * j;
    f.u[i] = *(const unsigned int*)(A + (size_t)m * lda + k0);
  }
  return f.v;
}

// B-matrix 32x16 bf16 fragment from an N-major (transposed) weight:
// lane L: n = L%16, kgrp = L/16; VGPR i: k0 = (i/4)*16 + kgrp*8 + 2*(i%4)  (pair packed per dword)
__device__ __forceinline__ v16bf load_frag_b(const __bf16* Bt, int ldb, int lane) {
  FragU f;
  const int n = lane & 15, kg = lane >> 4;
#pragma unroll
  for (int i = 0; i < 8; ++i) {
    const int k0 = (i >> 2) * 16 + kg * 8 + 2 * (i & 3);
    f.u[i] = *(const unsigned int*)(Bt + (size_t)n * ldb + k0);
  }
  return f.v;
}

__device__ __forceinline__ float sigmoid_f(float x) { return 1.f / (1.f + __expf(-x)); }
__device__ __forceinline__ float tanh_f(float x)    { return 1.f - 2.f / (__expf(2.f * x) + 1.f); }

// ---------------------------------------------------------------- converts
__global__ void f32_to_bf16_kernel(const float* __restrict__ in, __bf16* __restrict__ out, int n) {
  int i = blockIdx.x * blockDim.x + threadIdx.x;
  if (i < n) out[i] = (__bf16)in[i];
}

// in: (LAYERS, K, N) f32  ->  out: (LAYERS, N, K) bf16
__global__ void transpose_to_bf16_kernel(const float* __restrict__ in, __bf16* __restrict__ out,
                                         int K, int N) {
  int i = blockIdx.x * blockDim.x + threadIdx.x;
  int total = LAYERS * K * N;
  if (i >= total) return;
  int l = i / (K * N);
  int rem = i - l * (K * N);
  int k = rem / N;
  int n = rem - k * N;
  out[(size_t)l * N * K + (size_t)n * K + k] = (__bf16)in[i];
}

// ---------------------------------------------------------------- gx = LN(x @ wx)*g + b
// One block = 16 rows x 1024 cols. 512 threads = 16 waves; wave w covers
// cols [64w, 64w+64) -> 4 N-tiles. Low per-wave pressure + 4 waves/SIMD.
#define GX_THREADS 512
__global__ __launch_bounds__(GX_THREADS)
void gx_gemm_ln_kernel(const __bf16* __restrict__ xbf,  // (S*B, IN) bf16 row-major
                       const __bf16* __restrict__ wxT,  // (G4, IN) bf16  (N-major)
                       const float*  __restrict__ gamma,// 4H
                       const float*  __restrict__ beta, // 4H
                       float*        __restrict__ gx)   // (S*B, G4)
{
  __shared__ float s_sum[16];
  __shared__ float s_sqs[16];
  __shared__ float s_mv[16][2];

  const int tid  = threadIdx.x;
  const int wave = tid >> 5;
  const int lane = tid & 31;
  const int row0 = blockIdx.x * 16;
  const int mh   = lane >> 4;

  if (tid < 16) { s_sum[tid] = 0.f; s_sqs[tid] = 0.f; }
  __syncthreads();

  v8f acc[4];
#pragma unroll
  for (int nt = 0; nt < 4; ++nt) acc[nt] = (v8f){0.f,0.f,0.f,0.f,0.f,0.f,0.f,0.f};

#pragma unroll
  for (int kb = 0; kb < INDIM; kb += 32) {
    v16bf a = load_frag_a(xbf + (size_t)row0 * INDIM + kb, INDIM, lane);
    v16bf b[4];
#pragma unroll
    for (int nt = 0; nt < 4; ++nt) {
      const int nb = (wave * 4 + nt) * 16;
      b[nt] = load_frag_b(wxT + (size_t)nb * INDIM + kb, INDIM, lane);
    }
#pragma unroll
    for (int nt = 0; nt < 4; ++nt)
      acc[nt] = __builtin_amdgcn_wmma_f32_16x16x32_bf16(false, a, false, b[nt],
                                                        (short)0, acc[nt], false, false);
  }

  // row-wise LN statistics: shfl butterfly across the 16-lane N-group, then
  // only 2 lanes per wave touch the LDS accumulators.
#pragma unroll
  for (int r = 0; r < 8; ++r) {
    float s = 0.f, q = 0.f;
#pragma unroll
    for (int nt = 0; nt < 4; ++nt) { float v = acc[nt][r]; s += v; q += v * v; }
#pragma unroll
    for (int off = 8; off > 0; off >>= 1) {
      s += __shfl_xor(s, off, 32);
      q += __shfl_xor(q, off, 32);
    }
    if ((lane & 15) == 0) {
      atomicAdd(&s_sum[r + 8 * mh], s);
      atomicAdd(&s_sqs[r + 8 * mh], q);
    }
  }
  __syncthreads();
  if (tid < 16) {
    float mean = s_sum[tid] * (1.f / G4);
    float var  = s_sqs[tid] * (1.f / G4) - mean * mean;
    s_mv[tid][0] = mean;
    s_mv[tid][1] = rsqrtf(var + EPSV);
  }
  __syncthreads();

#pragma unroll
  for (int nt = 0; nt < 4; ++nt) {
    const int col = (wave * 4 + nt) * 16 + (lane & 15);
    const float gmm = gamma[col], btt = beta[col];
#pragma unroll
    for (int r = 0; r < 8; ++r) {
      const int mm = r + 8 * mh;
      float v = (acc[nt][r] - s_mv[mm][0]) * s_mv[mm][1];
      // gx is a write-once 134MB stream: keep it out of L2's hot set.
      __builtin_nontemporal_store(v * gmm + btt, &gx[(size_t)(row0 + mm) * G4 + col]);
    }
  }
}

// ---------------------------------------------------------------- persistent LSTM scan
// Block = 16 batch rows (independent recurrence), 1024 threads = 32 waves,
// wave w covers gate cols [32w, 32w+32) -> 2 N-tiles, 8 K-steps per t.
// 8 waves/SIMD hide the L2 latency of the (L2-resident) whT fragment loads.
#define SCAN_THREADS 1024
__global__ __launch_bounds__(SCAN_THREADS)
void lstm_scan_kernel(const __bf16* __restrict__ whT,   // (G4, H) bf16 N-major
                      const float*  __restrict__ gx,    // (S, B, G4)
                      const float*  __restrict__ bias,  // 4H
                      const float*  __restrict__ gamma, // gh gamma 4H
                      const float*  __restrict__ beta,  // gh beta 4H
                      const float*  __restrict__ h0,    // (B,H) for this layer
                      const float*  __restrict__ c0,
                      float*        __restrict__ ys,    // (S,B,H) or null
                      __bf16*       __restrict__ ysbf,  // (S,B,H) bf16 or null
                      float*        __restrict__ hT,    // (B,H)
                      float*        __restrict__ cT)
{
  __shared__ __bf16 s_h[16][HDIM];   // 8 KB  (A operand for next step)
  __shared__ float  s_c[16][HDIM];   // 16 KB
  __shared__ __bf16 s_g[16][G4];     // 32 KB gate exchange (raw h@wh)
  __shared__ float  s_sum[16], s_sqs[16];
  __shared__ float  s_mv[16][2];

  const int tid  = threadIdx.x;
  const int wave = tid >> 5;
  const int lane = tid & 31;
  const int b0   = blockIdx.x * 16;
  const int mh   = lane >> 4;

  for (int idx = tid; idx < 16 * HDIM; idx += SCAN_THREADS) {
    int r = idx >> 8, cc = idx & (HDIM - 1);
    s_h[r][cc] = (__bf16)h0[(size_t)(b0 + r) * HDIM + cc];
    s_c[r][cc] = c0[(size_t)(b0 + r) * HDIM + cc];
  }
  if (tid < 16) { s_sum[tid] = 0.f; s_sqs[tid] = 0.f; }
  __syncthreads();

  for (int t = 0; t < SEQ; ++t) {
    v8f acc[2];
#pragma unroll
    for (int nt = 0; nt < 2; ++nt) acc[nt] = (v8f){0.f,0.f,0.f,0.f,0.f,0.f,0.f,0.f};

#pragma unroll
    for (int kb = 0; kb < HDIM; kb += 32) {
      v16bf a = load_frag_a(&s_h[0][0] + kb, HDIM, lane);
      v16bf b[2];
#pragma unroll
      for (int nt = 0; nt < 2; ++nt) {
        const int nb = (wave * 2 + nt) * 16;
        b[nt] = load_frag_b(whT + (size_t)nb * HDIM + kb, HDIM, lane);
      }
#pragma unroll
      for (int nt = 0; nt < 2; ++nt)
        acc[nt] = __builtin_amdgcn_wmma_f32_16x16x32_bf16(false, a, false, b[nt],
                                                          (short)0, acc[nt], false, false);
    }

    // LN statistics (shfl butterfly + 2 atomic lanes/wave), publish raw gates
#pragma unroll
    for (int r = 0; r < 8; ++r) {
      float s = 0.f, q = 0.f;
#pragma unroll
      for (int nt = 0; nt < 2; ++nt) { float v = acc[nt][r]; s += v; q += v * v; }
#pragma unroll
      for (int off = 8; off > 0; off >>= 1) {
        s += __shfl_xor(s, off, 32);
        q += __shfl_xor(q, off, 32);
      }
      if ((lane & 15) == 0) {
        atomicAdd(&s_sum[r + 8 * mh], s);
        atomicAdd(&s_sqs[r + 8 * mh], q);
      }
    }
#pragma unroll
    for (int nt = 0; nt < 2; ++nt) {
      const int col = (wave * 2 + nt) * 16 + (lane & 15);
#pragma unroll
      for (int r = 0; r < 8; ++r)
        s_g[r + 8 * mh][col] = (__bf16)acc[nt][r];
    }
    __syncthreads();
    if (tid < 16) {
      float mean = s_sum[tid] * (1.f / G4);
      float var  = s_sqs[tid] * (1.f / G4) - mean * mean;
      s_mv[tid][0] = mean;
      s_mv[tid][1] = rsqrtf(var + EPSV);
    }
    __syncthreads();

    // re-zero the LN accumulators for step t+1 (sums were consumed above;
    // next step's atomics only start after the barrier at loop end)
    if (tid < 16) { s_sum[tid] = 0.f; s_sqs[tid] = 0.f; }

    // pointwise cell update
    const float* gxrow = gx + ((size_t)t * BATCH + b0) * G4;
    for (int idx = tid; idx < 16 * HDIM; idx += SCAN_THREADS) {
      const int rr = idx >> 8;
      const int hc = idx & (HDIM - 1);
      const float mean = s_mv[rr][0], rstd = s_mv[rr][1];
      const float* gxp = gxrow + (size_t)rr * G4;

      float gi = ((float)s_g[rr][hc]            - mean) * rstd * gamma[hc]            + beta[hc]
               + __builtin_nontemporal_load(&gxp[hc])            + bias[hc];
      float gf = ((float)s_g[rr][HDIM + hc]     - mean) * rstd * gamma[HDIM + hc]     + beta[HDIM + hc]
               + __builtin_nontemporal_load(&gxp[HDIM + hc])     + bias[HDIM + hc];
      float go = ((float)s_g[rr][2 * HDIM + hc] - mean) * rstd * gamma[2 * HDIM + hc] + beta[2 * HDIM + hc]
               + __builtin_nontemporal_load(&gxp[2 * HDIM + hc]) + bias[2 * HDIM + hc];
      float gg = ((float)s_g[rr][3 * HDIM + hc] - mean) * rstd * gamma[3 * HDIM + hc] + beta[3 * HDIM + hc]
               + __builtin_nontemporal_load(&gxp[3 * HDIM + hc]) + bias[3 * HDIM + hc];

      float iv = sigmoid_f(gi);
      float fv = sigmoid_f(gf);
      float ov = sigmoid_f(go);
      float gv = tanh_f(gg);

      float cn = fv * s_c[rr][hc] + iv * gv;
      float hn = ov * tanh_f(cn);

      s_c[rr][hc] = cn;
      s_h[rr][hc] = (__bf16)hn;

      const size_t oidx = ((size_t)t * BATCH + b0 + rr) * HDIM + hc;
      if (ys)   __builtin_nontemporal_store(hn, &ys[oidx]);
      if (ysbf) ysbf[oidx] = (__bf16)hn;
      if (t == SEQ - 1) {
        hT[(size_t)(b0 + rr) * HDIM + hc] = hn;
        cT[(size_t)(b0 + rr) * HDIM + hc] = cn;
      }
    }
    __syncthreads();
  }
}

// ---------------------------------------------------------------- host
extern "C" void kernel_launch(void* const* d_in, const int* in_sizes, int n_in,
                              void* d_out, int out_size, void* d_ws, size_t ws_size,
                              hipStream_t stream) {
  const float* x    = (const float*)d_in[0];
  const float* h0   = (const float*)d_in[1];
  const float* c0   = (const float*)d_in[2];
  const float* wx   = (const float*)d_in[3];
  const float* wh   = (const float*)d_in[4];
  const float* bias = (const float*)d_in[5];
  const float* lng  = (const float*)d_in[6];
  const float* lnb  = (const float*)d_in[7];
  float* out = (float*)d_out;

  char* ws = (char*)d_ws;
  size_t off = 0;
  auto wsalloc = [&](size_t bytes) -> void* {
    void* p = ws + off;
    off += (bytes + 255) & ~(size_t)255;
    return p;
  };
  __bf16* xbf  = (__bf16*)wsalloc((size_t)SEQ * BATCH * INDIM * 2);
  __bf16* ysbf = (__bf16*)wsalloc((size_t)SEQ * BATCH * HDIM * 2);
  __bf16* wxT  = (__bf16*)wsalloc((size_t)LAYERS * G4 * INDIM * 2);
  __bf16* whT  = (__bf16*)wsalloc((size_t)LAYERS * G4 * HDIM * 2);
  float*  gxb  = (float*)wsalloc((size_t)SEQ * BATCH * G4 * 4);

  float* xout  = out;
  float* hsout = out + (size_t)SEQ * BATCH * HDIM;
  float* csout = hsout + (size_t)LAYERS * BATCH * HDIM;

  const int nx = SEQ * BATCH * INDIM;
  f32_to_bf16_kernel<<<(nx + 255) / 256, 256, 0, stream>>>(x, xbf, nx);
  const int nw = LAYERS * INDIM * G4;
  transpose_to_bf16_kernel<<<(nw + 255) / 256, 256, 0, stream>>>(wx, wxT, INDIM, G4);
  transpose_to_bf16_kernel<<<(nw + 255) / 256, 256, 0, stream>>>(wh, whT, HDIM, G4);

  for (int l = 0; l < LAYERS; ++l) {
    const __bf16* xin = (l == 0) ? xbf : ysbf;
    gx_gemm_ln_kernel<<<SEQ * BATCH / 16, GX_THREADS, 0, stream>>>(
        xin, wxT + (size_t)l * G4 * INDIM,
        lng + (size_t)l * 8 * HDIM, lnb + (size_t)l * 8 * HDIM, gxb);

    float*  ysl   = (l == LAYERS - 1) ? xout : (float*)nullptr;
    __bf16* ysbfl = (l == LAYERS - 1) ? (__bf16*)nullptr : ysbf;
    lstm_scan_kernel<<<BATCH / 16, SCAN_THREADS, 0, stream>>>(
        whT + (size_t)l * G4 * HDIM, gxb, bias + (size_t)l * G4,
        lng + (size_t)l * 8 * HDIM + G4, lnb + (size_t)l * 8 * HDIM + G4,
        h0 + (size_t)l * BATCH * HDIM, c0 + (size_t)l * BATCH * HDIM,
        ysl, ysbfl,
        hsout + (size_t)l * BATCH * HDIM, csout + (size_t)l * BATCH * HDIM);
  }
}
